// GloAttnConv_76038101009025
// MI455X (gfx1250) — matrix operand order repaired and checked by reading, not compiled, not run on hardware
//
#include <hip/hip_runtime.h>
#include <hip/hip_bf16.h>

typedef __attribute__((ext_vector_type(16))) _Float16 v16h;
typedef __attribute__((ext_vector_type(8)))  _Float16 v8h;
typedef __attribute__((ext_vector_type(8)))  float    v8f;

#define NODES   65536
#define NPG     4096
#define NGRAPH  16
#define NHEAD   2
#define DIM     64
#define EDGES   524288
#define BETA    0.5f

// ---------------- WMMA helpers (gfx1250, wave32) ----------------

__device__ __forceinline__ v8f wmma16x16x32(v16h a, v16h b, v8f c) {
    return __builtin_amdgcn_wmma_f32_16x16x32_f16(
        /*neg_a=*/false, a, /*neg_b=*/false, b,
        /*c_mod=*/(short)0, c, /*reuse_a=*/false, /*reuse_b=*/false);
}

// Async global->LDS copy (CDNA5 GLOBAL_LOAD_ASYNC_TO_LDS_B128, ASYNCcnt-tracked).
// LDS byte offset = low 32 bits of the generic address of a __shared__ object.
__device__ __forceinline__ void async_copy_b128(void* lds_ptr, const void* gptr) {
    unsigned lds_off = (unsigned)(size_t)lds_ptr;
    asm volatile("global_load_async_to_lds_b128 %0, %1, off"
                 :: "v"(lds_off), "v"((unsigned long long)(size_t)gptr)
                 : "memory");
}
__device__ __forceinline__ void wait_async0() {
    asm volatile("s_wait_asynccnt 0" ::: "memory");
}

// A-fragment (16x32, f16, row-major source fp32). Layout per ISA:
// element j of lane L -> k = (j>>3)*16 + (L>>4)*8 + (j&7), m = L&15.
// => two contiguous 8-float chunks per lane at k0 = (L>>4)*8 and k0+16.
__device__ __forceinline__ v16h load_a_f32(const float* base, int ld, int lane) {
    int m = lane & 15, g = lane >> 4;
    const float* p = base + (size_t)m * ld + g * 8;
    float4 f0 = *(const float4*)(p);
    float4 f1 = *(const float4*)(p + 4);
    float4 f2 = *(const float4*)(p + 16);
    float4 f3 = *(const float4*)(p + 20);
    v16h a;
    a[0]=(_Float16)f0.x; a[1]=(_Float16)f0.y; a[2]=(_Float16)f0.z; a[3]=(_Float16)f0.w;
    a[4]=(_Float16)f1.x; a[5]=(_Float16)f1.y; a[6]=(_Float16)f1.z; a[7]=(_Float16)f1.w;
    a[8]=(_Float16)f2.x; a[9]=(_Float16)f2.y; a[10]=(_Float16)f2.z; a[11]=(_Float16)f2.w;
    a[12]=(_Float16)f3.x; a[13]=(_Float16)f3.y; a[14]=(_Float16)f3.z; a[15]=(_Float16)f3.w;
    return a;
}

// A-fragment from f16 row-major source: two 16-byte contiguous chunks per lane.
__device__ __forceinline__ v16h load_a_f16(const _Float16* base, int ld, int lane) {
    int m = lane & 15, g = lane >> 4;
    const _Float16* p = base + (size_t)m * ld + g * 8;
    v8h lo = *(const v8h*)(p);
    v8h hi = *(const v8h*)(p + 16);
    v16h a;
#pragma unroll
    for (int i = 0; i < 8; ++i) { a[i] = lo[i]; a[i + 8] = hi[i]; }
    return a;
}

// Packed B-fragment: frag f stored as 32 lanes x 16 halves contiguous.
__device__ __forceinline__ v16h load_b_packed(const _Float16* packed, int frag, int lane) {
    return *(const v16h*)(packed + (((size_t)frag * 32 + lane) << 4));
}

// ---------------- pack kernels ----------------

// B layout: element j of lane L -> k = (j>>3)*16 + (L>>4)*8 + (j&7), n = L&15 (per 32x16 tile).
__global__ void k_packW(const float* __restrict__ W, int K, int Nn,
                        _Float16* __restrict__ out, int total) {
    int idx = blockIdx.x * blockDim.x + threadIdx.x;
    if (idx >= total) return;
    int frag = idx >> 9, lane = (idx >> 4) & 31, j = idx & 15;
    int ntiles = Nn >> 4;
    int kt = frag / ntiles, nt = frag % ntiles;
    int k = kt * 32 + ((j >> 3) << 4) + ((lane >> 4) << 3) + (j & 7);
    int n = nt * 16 + (lane & 15);
    out[idx] = (_Float16)W[(size_t)k * Nn + n];
}

// kvs_ext per (b,h): K=64, N=80 (cols 0..63 = kvs, 64..79 = ksum replicated)
__global__ void k_packKvs(const float* __restrict__ kvs, const float* __restrict__ ksum,
                          _Float16* __restrict__ out) {
    int idx = blockIdx.x * blockDim.x + threadIdx.x;
    if (idx >= NGRAPH * NHEAD * 10 * 512) return;
    int bh = idx / 5120, rem = idx % 5120;
    int frag = rem >> 9, lane = (rem >> 4) & 31, j = rem & 15;
    int kt = frag / 5, nt = frag % 5;
    int k = kt * 32 + ((j >> 3) << 4) + ((lane >> 4) << 3) + (j & 7);
    float v = (nt < 4) ? kvs[(size_t)bh * 4096 + (size_t)k * 64 + nt * 16 + (lane & 15)]
                       : ksum[(size_t)bh * 64 + k];
    out[idx] = (_Float16)v;
}

// ---------------- projection: q = xWq+bq, k = xWk+bk ----------------
// block = 128 (4 waves); wave handles 4 of the 16 output 16-col tiles (8 for q, 8 for k)
__global__ void k_proj(const float* __restrict__ x,
                       const _Float16* __restrict__ WqP, const _Float16* __restrict__ WkP,
                       const float* __restrict__ bq, const float* __restrict__ bk,
                       float* __restrict__ q, float* __restrict__ k) {
    int lane = threadIdx.x & 31, wave = threadIdx.x >> 5;
    int row0 = blockIdx.x * 16;
    v16h a0 = load_a_f32(x + (size_t)row0 * DIM, DIM, lane);
    v16h a1 = load_a_f32(x + (size_t)row0 * DIM + 32, DIM, lane);
#pragma unroll
    for (int t = 0; t < 4; ++t) {
        int nt = wave * 4 + t;
        bool isQ = nt < 8;
        int ct = nt & 7;
        const _Float16* Wp = isQ ? WqP : WkP;
        v16h b0 = load_b_packed(Wp, 0 * 8 + ct, lane);
        v16h b1 = load_b_packed(Wp, 1 * 8 + ct, lane);
        v8f c = {0.f,0.f,0.f,0.f,0.f,0.f,0.f,0.f};
        c = wmma16x16x32(a0, b0, c);
        c = wmma16x16x32(a1, b1, c);
        int col = ct * 16 + (lane & 15);
        float bias = (isQ ? bq : bk)[col];
        float* out = isQ ? q : k;
#pragma unroll
        for (int v = 0; v < 8; ++v) {
            int row = row0 + v + ((lane >> 4) << 3);
            out[(size_t)row * 128 + col] = c[v] + bias;
        }
    }
}

// ---------------- per-(node,head) L2 normalize -> f16 ----------------
__global__ void k_norm(const float* __restrict__ q, const float* __restrict__ k,
                       _Float16* __restrict__ qs, _Float16* __restrict__ ks) {
    int lane = threadIdx.x & 31;
    int wid = blockIdx.x * 8 + (threadIdx.x >> 5);
    int n = wid >> 1, h = wid & 1;
    size_t base = (size_t)n * 128 + (size_t)h * 64;
    size_t ob = ((size_t)n * NHEAD + h) * DIM;

    float a = q[base + lane], b = q[base + lane + 32];
    float ss = a * a + b * b;
#pragma unroll
    for (int o = 16; o > 0; o >>= 1) ss += __shfl_xor(ss, o, 32);
    float r = rsqrtf(ss + 1e-24f);
    qs[ob + lane] = (_Float16)(a * r);
    qs[ob + lane + 32] = (_Float16)(b * r);

    a = k[base + lane]; b = k[base + lane + 32];
    ss = a * a + b * b;
#pragma unroll
    for (int o = 16; o > 0; o >>= 1) ss += __shfl_xor(ss, o, 32);
    r = rsqrtf(ss + 1e-24f);
    ks[ob + lane] = (_Float16)(a * r);
    ks[ob + lane + 32] = (_Float16)(b * r);
}

// ---------------- misc small kernels ----------------
__global__ void k_zero(float* __restrict__ p, int n) {
    int i = blockIdx.x * blockDim.x + threadIdx.x;
    if (i < n) p[i] = 0.f;
}
__global__ void k_deg(const int* __restrict__ ei, float* __restrict__ deg) {
    int e = blockIdx.x * blockDim.x + threadIdx.x;
    if (e < EDGES) atomicAdd(&deg[ei[EDGES + e]], 1.0f);
}
__global__ void k_dinv(const float* __restrict__ deg, float* __restrict__ dinv) {
    int n = blockIdx.x * blockDim.x + threadIdx.x;
    if (n < NODES) { float d = deg[n]; dinv[n] = d > 0.f ? rsqrtf(d) : 0.f; }
}
__global__ void k_initxs(const float* __restrict__ x, float* __restrict__ xs) {
    int i = blockIdx.x * blockDim.x + threadIdx.x;   // N*128
    int n = i >> 7, c = i & 127;
    xs[(size_t)i] = x[(size_t)n * DIM + (c & 63)];
}
__global__ void k_combine(const float* __restrict__ xs, const float* __restrict__ attn,
                          float* __restrict__ acc) {
    size_t i = (size_t)blockIdx.x * blockDim.x + threadIdx.x;  // N*128
    acc[i] = xs[i] + (1.0f - BETA) * attn[i];
}
__global__ void k_scatter(const int* __restrict__ ei, const float* __restrict__ dinv,
                          const float* __restrict__ xs, float* __restrict__ acc) {
    int t = threadIdx.x;
    int e = blockIdx.x * 2 + (t >> 7);
    int c = t & 127;
    int r = ei[e], cl = ei[EDGES + e];
    // hide the latency of the gathered row behind the dinv loads
    __builtin_prefetch(&xs[(size_t)r * 128 + c], 0, 1);
    float w = BETA * dinv[r] * dinv[cl];
    atomicAdd(&acc[(size_t)cl * 128 + c], w * xs[(size_t)r * 128 + c]);
}

// ---------------- kvs / ksum / xsum reduction ----------------
// grid = B*H*8 slices of 512 nodes; stages 32-node chunks of x (fp32) and ks (f16)
// into LDS with CDNA5 async global->LDS copies, then does an outer-product
// accumulation from LDS.
__global__ void k_kvs(const float* __restrict__ x, const _Float16* __restrict__ ks,
                      float* __restrict__ kvs, float* __restrict__ ksum,
                      float* __restrict__ xsum) {
    __shared__ float    xls[32 * 64];
    __shared__ _Float16 kls[32 * 64];
    int t = threadIdx.x;
    int b = blockIdx.x / (NHEAD * 8);
    int rem = blockIdx.x % (NHEAD * 8);
    int h = rem / 8, slice = rem % 8;
    int nodeStart = b * NPG + slice * 512;
    int bh = b * NHEAD + h;

    int m = t >> 2, db = (t & 3) * 16;
    float acc[16];
#pragma unroll
    for (int i = 0; i < 16; ++i) acc[i] = 0.f;
    float kacc = 0.f, xacc = 0.f;

    int l = t >> 3, sub = t & 7;       // 32 nodes x 8 sub-chunks
    for (int chunk = 0; chunk < 16; ++chunk) {
        int base = nodeStart + chunk * 32;
        int node = base + l;
        // x row: 256B -> 2 x b128 per thread; ks row (this head): 128B -> 1 x b128
        const float* gx = x + (size_t)node * DIM + sub * 8;
        async_copy_b128(&xls[l * 64 + sub * 8], gx);
        async_copy_b128(&xls[l * 64 + sub * 8 + 4], gx + 4);
        const _Float16* gk = ks + ((size_t)node * NHEAD + h) * DIM + sub * 8;
        async_copy_b128(&kls[l * 64 + sub * 8], gk);
        wait_async0();
        __syncthreads();
#pragma unroll 4
        for (int ll = 0; ll < 32; ++ll) {
            float kv = (float)kls[ll * 64 + m];
            const float* xr = &xls[ll * 64 + db];
#pragma unroll
            for (int dd = 0; dd < 16; ++dd) acc[dd] = fmaf(kv, xr[dd], acc[dd]);
        }
        if (t < 64) {
            for (int ll = 0; ll < 32; ++ll) kacc += (float)kls[ll * 64 + t];
        } else if (t < 128 && h == 0) {
            int d = t - 64;
            for (int ll = 0; ll < 32; ++ll) xacc += xls[ll * 64 + d];
        }
        __syncthreads();
    }
#pragma unroll
    for (int dd = 0; dd < 16; ++dd)
        atomicAdd(&kvs[(size_t)bh * 4096 + (size_t)m * 64 + db + dd], acc[dd]);
    if (t < 64) atomicAdd(&ksum[(size_t)bh * 64 + t], kacc);
    else if (t < 128 && h == 0) atomicAdd(&xsum[(size_t)b * 64 + (t - 64)], xacc);
}

// ---------------- attention: qs @ [kvs | ksum_rep] with fused divide ----------------
// block = 128 (4 waves); each wave owns one 16-row tile of one (b,h).
__global__ void k_attn(const _Float16* __restrict__ qs, const _Float16* __restrict__ kvsP,
                       const float* __restrict__ xsum, float* __restrict__ attn) {
    int lane = threadIdx.x & 31;
    int waveG = blockIdx.x * 4 + (threadIdx.x >> 5);
    int bh = waveG >> 8;          // 256 tiles per (b,h)
    int tile = waveG & 255;
    int b = bh >> 1, h = bh & 1;
    int node0 = b * NPG + tile * 16;

    const _Float16* abase = qs + ((size_t)node0 * NHEAD + h) * DIM;
    v16h a0 = load_a_f16(abase, NHEAD * DIM, lane);
    v16h a1 = load_a_f16(abase + 32, NHEAD * DIM, lane);
    const _Float16* bp = kvsP + (size_t)bh * 5120;

    v8f c[5];
#pragma unroll
    for (int nt = 0; nt < 5; ++nt) {
        v8f acc = {0.f,0.f,0.f,0.f,0.f,0.f,0.f,0.f};
        acc = wmma16x16x32(a0, load_b_packed(bp, 0 * 5 + nt, lane), acc);
        acc = wmma16x16x32(a1, load_b_packed(bp, 1 * 5 + nt, lane), acc);
        c[nt] = acc;
    }
    // c[4] holds den = qs . ksum (same row for every lane); +B = 16 graphs.
#pragma unroll
    for (int nt = 0; nt < 4; ++nt) {
        int col = nt * 16 + (lane & 15);
        float xa = xsum[(size_t)b * 64 + col];
#pragma unroll
        for (int v = 0; v < 8; ++v) {
            int node = node0 + v + ((lane >> 4) << 3);
            attn[((size_t)node * NHEAD + h) * DIM + col] =
                (c[nt][v] + xa) / (c[4][v] + (float)NGRAPH);
        }
    }
}

// ---------------- output projection: (xs @ Wo + bo) / H ----------------
__global__ void k_out(const float* __restrict__ xs, const _Float16* __restrict__ WoP,
                      const float* __restrict__ bo, float* __restrict__ out) {
    int lane = threadIdx.x & 31, nt = threadIdx.x >> 5;  // 4 waves, 4 col-tiles
    int row0 = blockIdx.x * 16;
    v8f c = {0.f,0.f,0.f,0.f,0.f,0.f,0.f,0.f};
#pragma unroll
    for (int kt = 0; kt < 4; ++kt) {
        v16h a = load_a_f32(xs + (size_t)row0 * 128 + kt * 32, 128, lane);
        v16h b = load_b_packed(WoP, kt * 4 + nt, lane);
        c = wmma16x16x32(a, b, c);
    }
    int col = nt * 16 + (lane & 15);
    float bias = bo[col];
#pragma unroll
    for (int v = 0; v < 8; ++v) {
        int row = row0 + v + ((lane >> 4) << 3);
        out[(size_t)row * 64 + col] = (c[v] + bias) * (1.0f / NHEAD);
    }
}

// ---------------- host launch ----------------

extern "C" void kernel_launch(void* const* d_in, const int* in_sizes, int n_in,
                              void* d_out, int out_size, void* d_ws, size_t ws_size,
                              hipStream_t stream) {
    const float* x  = (const float*)d_in[0];
    const int*   ei = (const int*)d_in[1];
    const float* Wq = (const float*)d_in[3];
    const float* bq = (const float*)d_in[4];
    const float* Wk = (const float*)d_in[5];
    const float* bk = (const float*)d_in[6];
    const float* Wo = (const float*)d_in[7];
    const float* bo = (const float*)d_in[8];
    float* out = (float*)d_out;

    char* w = (char*)d_ws;
    // 32MB q fp32 (reused as xs buffer A), 32MB k fp32 (reused as xs buffer B)
    float* bufQ = (float*)(w + 0);
    float* bufK = (float*)(w + 33554432ull);
    float* attn = (float*)(w + 67108864ull);
    _Float16* qsF = (_Float16*)(w + 100663296ull);
    _Float16* ksF = (_Float16*)(w + 117440512ull);
    float* deg  = (float*)(w + 134217728ull);          // 256KB
    float* kvs  = (float*)(w + 134479872ull);          // 512KB
    float* ksum = (float*)(w + 135004160ull);          // 8KB
    float* xsum = (float*)(w + 135012352ull);          // 4KB
    float* dinv = (float*)(w + 135016448ull);          // 256KB
    _Float16* WqP  = (_Float16*)(w + 135278592ull);    // 16KB
    _Float16* WkP  = (_Float16*)(w + 135294976ull);    // 16KB
    _Float16* WoP  = (_Float16*)(w + 135311360ull);    // 16KB
    _Float16* kvsP = (_Float16*)(w + 135327744ull);    // 320KB

    // 1) pack weights to f16 B-fragments
    k_packW<<<32, 256, 0, stream>>>(Wq, 64, 128, WqP, 8192);
    k_packW<<<32, 256, 0, stream>>>(Wk, 64, 128, WkP, 8192);
    k_packW<<<32, 256, 0, stream>>>(Wo, 128, 64, WoP, 8192);

    // 2) q/k projection (WMMA)
    k_proj<<<NODES / 16, 128, 0, stream>>>(x, WqP, WkP, bq, bk, bufQ, bufK);

    // 3) normalize -> qs/ks f16
    k_norm<<<NODES * NHEAD / 8, 256, 0, stream>>>(bufQ, bufK, qsF, ksF);

    // 4) zero accumulators (deg|kvs|ksum|xsum contiguous = 199680 floats)
    k_zero<<<(199680 + 255) / 256, 256, 0, stream>>>(deg, 199680);
    k_deg<<<EDGES / 256, 256, 0, stream>>>(ei, deg);
    k_dinv<<<NODES / 256, 256, 0, stream>>>(deg, dinv);

    // 5) kvs/ksum/xsum reduction (async global->LDS staging)
    k_kvs<<<NGRAPH * NHEAD * 8, 256, 0, stream>>>(x, ksF, kvs, ksum, xsum);
    k_packKvs<<<(163840 + 255) / 256, 256, 0, stream>>>(kvs, ksum, kvsP);

    // 6) attention (WMMA, fused num/den)
    k_attn<<<NGRAPH * NHEAD * 256 / 4, 128, 0, stream>>>(qsF, kvsP, xsum, attn);

    // 7) xs init + 4 propagation iterations (q/k fp32 buffers reused as double buffer)
    float* xs = bufQ;
    float* acc = bufK;
    k_initxs<<<NODES * 128 / 256, 256, 0, stream>>>(x, xs);
    for (int it = 0; it < 4; ++it) {
        k_combine<<<NODES * 128 / 256, 256, 0, stream>>>(xs, attn, acc);
        k_scatter<<<EDGES / 2, 256, 0, stream>>>(ei, dinv, xs, acc);
        float* tmp = xs; xs = acc; acc = tmp;
    }

    // 8) output projection (WMMA)
    k_out<<<NODES / 16, 128, 0, stream>>>(xs, WoP, bo, out);
}
